// IMG2TEXTLOCAL_49014166782024
// MI455X (gfx1250) — matrix operand order repaired and checked by reading, not compiled, not run on hardware
//
#include <hip/hip_runtime.h>
#include <math.h>

typedef __attribute__((ext_vector_type(16))) _Float16 v16h;
typedef __attribute__((ext_vector_type(8)))  _Float16 v8h;
typedef __attribute__((ext_vector_type(8)))  float    v8f;

#define BB   128
#define PP   197
#define DD   768
#define TOKN 512
#define KK   16
#define TOPKN 8
#define LL   2
#define HH   12
#define HD   64
#define NTOK 213   /* K + P */
#define NP   224   /* padded tokens */
#define MROWS (BB*NP)
#define EPSV 0.05f

// ---------------------------------------------------------------- async-to-LDS (CDNA5)
// Inline asm: GLOBAL_LOAD_ASYNC_TO_LDS_B128 (ASYNCcnt-tracked DMA into LDS).
static __device__ inline void cp16_async(void* lds, const void* g) {
  unsigned laddr = (unsigned)(size_t)(__attribute__((address_space(3))) char*)lds;
  asm volatile("global_load_async_to_lds_b128 %0, %1, off"
               :: "v"(laddr), "v"(g) : "memory");
}
static __device__ inline void async_wait0() {
  asm volatile("s_wait_asynccnt 0x0" ::: "memory");
}

// wave index as compiler-visible scalar (keeps WMMA control flow EXEC-clean)
static __device__ inline int wave_id() {
  return __builtin_amdgcn_readfirstlane((int)(threadIdx.x >> 5));
}

// ---------------------------------------------------------------- WMMA utils
static __device__ inline v8f wmma16(v16h a, v16h b, v8f c) {
  return __builtin_amdgcn_wmma_f32_16x16x32_f16(false, a, false, b, (short)0, c,
                                                false, false);
}

// A-matrix 16x32 fragment from LDS ([m][k] row-major, stride in halfs)
static __device__ inline v16h frag_a(const _Float16* p, int stride) {
  int lane = threadIdx.x & 31;
  const _Float16* r = p + (lane & 15) * stride + (lane >> 4) * 8;
  union { v16h v; v8h h[2]; } u;
  u.h[0] = *(const v8h*)(r);
  u.h[1] = *(const v8h*)(r + 16);
  return u.v;
}

// B-matrix 32x16 fragment from LDS stored N-major ([n][k], stride in halfs)
static __device__ inline v16h frag_b(const _Float16* p, int stride) {
  int lane = threadIdx.x & 31;
  const _Float16* r = p + (lane & 15) * stride + (lane >> 4) * 8;
  union { v16h v; v8h h[2]; } u;
  u.h[0] = *(const v8h*)(r);
  u.h[1] = *(const v8h*)(r + 16);
  return u.v;
}

// ---------------------------------------------------------------- misc kernels
__global__ void cvt_f16_kernel(const float* __restrict__ src,
                               _Float16* __restrict__ dst, long n) {
  long i = (long)blockIdx.x * 256 + threadIdx.x;
  long stride = (long)gridDim.x * 256;
  for (; i < n; i += stride) dst[i] = (_Float16)src[i];
}

__global__ void initx_kernel(const float* __restrict__ tpl,
                             const float* __restrict__ patches,
                             float* __restrict__ X, long n) {
  long i = (long)blockIdx.x * 256 + threadIdx.x;
  long stride = (long)gridDim.x * 256;
  for (; i < n; i += stride) {
    long d = i % DD;
    long row = i / DD;
    int tok = (int)(row % NP);
    long b = row / NP;
    float v;
    if (tok < KK)        v = tpl[(long)tok * DD + d];
    else if (tok < NTOK) v = patches[(b * PP + (tok - KK)) * DD + d];
    else                 v = 0.f;
    X[i] = v;
  }
}

__global__ void extract_kernel(const float* __restrict__ X,
                               _Float16* __restrict__ XT) {
  long i = (long)blockIdx.x * 256 + threadIdx.x;
  long n = (long)BB * KK * DD;
  if (i < n) {
    long d = i % DD;
    long r = i / DD;
    long tok = r % KK;
    long b = r / KK;
    XT[i] = (_Float16)X[(b * NP + tok) * DD + d];
  }
}

// ---------------------------------------------------------------- LayerNorm (f32 in, f16 out)
__global__ void ln_kernel(const float* __restrict__ X,
                          const float* __restrict__ w,
                          const float* __restrict__ bvec,
                          _Float16* __restrict__ Y) {
  __shared__ float red[256];
  __shared__ float red2[256];
  int row = blockIdx.x, tid = threadIdx.x;
  const float* x = X + (size_t)row * DD;
  float s = 0.f, s2 = 0.f;
  for (int i = tid; i < DD; i += 256) { float v = x[i]; s += v; s2 += v * v; }
  red[tid] = s; red2[tid] = s2;
  __syncthreads();
  for (int off = 128; off; off >>= 1) {
    if (tid < off) { red[tid] += red[tid + off]; red2[tid] += red2[tid + off]; }
    __syncthreads();
  }
  float mean = red[0] * (1.f / DD);
  float var  = red2[0] * (1.f / DD) - mean * mean;
  float inv  = rsqrtf(var + 1e-5f);
  _Float16* y = Y + (size_t)row * DD;
  for (int i = tid; i < DD; i += 256)
    y[i] = (_Float16)((x[i] - mean) * inv * w[i] + bvec[i]);
}

// ---------------------------------------------------------------- WMMA GEMM, 128x64 block tile
// C[M x N] = A[M x K] * W[K x N] (+bias); A,W f16 row-major. One wave per 16-row
// strip, one A-fragment reused across 4 B-fragments (4 wmma / wave / k-step).
// MODE 0: out16 = v   1: out16 = relu(v)   2: out32 += v   3: out32 = sigmoid(v)
template <int MODE>
__global__ void gemm_wmma_kernel(const _Float16* __restrict__ A, int lda,
                                 const _Float16* __restrict__ W, int ldw,
                                 const float* __restrict__ bias,
                                 _Float16* __restrict__ out16,
                                 float* __restrict__ out32,
                                 int ldo, int Kdim) {
  __shared__ _Float16 As[128 * 32];   // [m][k]
  __shared__ _Float16 Bs[64 * 32];    // [n][k] (transposed from W)
  int tid = threadIdx.x;
  int wave = wave_id(), lane = tid & 31;
  int blockN = blockIdx.x * 64, blockM = blockIdx.y * 128;
  v8f acc[4] = {{}, {}, {}, {}};
  for (int k0 = 0; k0 < Kdim; k0 += 32) {
    // A tile 128x32: 512 16B chunks, 2 per thread, async to LDS
    #pragma unroll
    for (int cc = 0; cc < 2; ++cc) {
      int c = tid + cc * 256;
      int row = c >> 2, col = (c & 3) * 8;
      cp16_async(&As[c * 8], A + (size_t)(blockM + row) * lda + k0 + col);
    }
    // B tile: coalesced read along N, transposed store to [n][k]
    #pragma unroll
    for (int j = 0; j < 8; ++j) {
      int e = tid + 256 * j;
      int n = e & 63, k = e >> 6;
      Bs[n * 32 + k] = W[(size_t)(k0 + k) * ldw + blockN + n];
    }
    async_wait0();
    __syncthreads();
    v16h a = frag_a(As + wave * 16 * 32, 32);
    #pragma unroll
    for (int nt = 0; nt < 4; ++nt) {
      v16h b = frag_b(Bs + nt * 16 * 32, 32);
      acc[nt] = wmma16(a, b, acc[nt]);
    }
    __syncthreads();
  }
  int mrow = (lane < 16) ? 0 : 8;
  int ncol = lane & 15;
  #pragma unroll
  for (int nt = 0; nt < 4; ++nt) {
    #pragma unroll
    for (int r = 0; r < 8; ++r) {
      int row = blockM + wave * 16 + mrow + r;
      int col = blockN + nt * 16 + ncol;
      float v = acc[nt][r] + bias[col];
      size_t o = (size_t)row * ldo + col;
      if (MODE == 0)      out16[o] = (_Float16)v;
      else if (MODE == 1) out16[o] = (_Float16)fmaxf(v, 0.f);
      else if (MODE == 2) out32[o] += v;
      else                out32[o] = 1.f / (1.f + expf(-v));
    }
  }
}

// ---------------------------------------------------------------- fused attention
// block = (query-tile qt of 32, head h, batch b); 256 threads = 8 waves.
__global__ void attn_kernel(const _Float16* __restrict__ Q,
                            const _Float16* __restrict__ KV,
                            _Float16* __restrict__ O) {
  extern __shared__ char smem[];
  _Float16* Qs = (_Float16*)smem;            // 32 x 64   [m][k]
  _Float16* Ks = Qs + 32 * 64;               // 224 x 64  [key][dim]
  _Float16* Vs = Ks + 224 * 64;              // 64 x 224  [dim][key]
  float*    Ss = (float*)(Vs + 64 * 224);    // 32 x 224  scores f32
  _Float16* Ps = (_Float16*)(Ss + 32 * 224); // 32 x 224  probs f16
  int tid = threadIdx.x;
  int wave = wave_id(), lane = tid & 31;
  int qt = blockIdx.x, h = blockIdx.y, b = blockIdx.z;
  size_t qbase  = (size_t)b * NP * DD + (size_t)h * HD;
  size_t kvbase = (size_t)b * NP * (2 * DD) + (size_t)h * HD;

  // Q tile: 256 16B chunks (one per thread), async
  {
    int m = tid >> 3, e = (tid & 7) * 8;
    cp16_async(&Qs[tid * 8], Q + qbase + (size_t)(qt * 32 + m) * DD + e);
  }
  // K tile: 1792 16B chunks (7 per thread), async
  #pragma unroll
  for (int cc = 0; cc < 7; ++cc) {
    int c = tid + cc * 256;
    int key = c >> 3, e = (c & 7) * 8;
    cp16_async(&Ks[c * 8], KV + kvbase + (size_t)key * (2 * DD) + e);
  }
  // V tile transposed: scalar (read coalesced, scattered LDS stores)
  for (int i = tid; i < 224 * 64; i += 256) {
    int key = i >> 6, e = i & 63;
    Vs[e * 224 + key] = KV[kvbase + (size_t)key * (2 * DD) + DD + e];
  }
  async_wait0();
  __syncthreads();

  // Phase 1: S = scale * Q K^T  (28 16x16 tiles, 2 wmma each over k=64)
  // 'wave' is an SGPR -> scalar loop, EXEC untouched around WMMA.
  for (int t = wave; t < 28; t += 8) {
    int mt = t / 14, nt = t % 14;
    v8f acc = {};
    #pragma unroll
    for (int kc = 0; kc < 2; ++kc) {
      v16h a  = frag_a(Qs + mt * 16 * 64 + kc * 32, 64);
      v16h bf = frag_b(Ks + nt * 16 * 64 + kc * 32, 64);
      acc = wmma16(a, bf, acc);
    }
    int mrow = (lane < 16) ? 0 : 8;
    int key  = nt * 16 + (lane & 15);
    #pragma unroll
    for (int r = 0; r < 8; ++r) {
      float v = (key < NTOK) ? acc[r] * 0.125f : -1e30f;   // hd^-0.5, mask pads
      Ss[(mt * 16 + mrow + r) * 224 + key] = v;
    }
  }
  __syncthreads();

  // Phase 2: row softmax (8 lanes per row, shuffle reduce)
  {
    int row = tid >> 3, j = tid & 7;
    float m = -1e30f;
    #pragma unroll
    for (int i = 0; i < 28; ++i) m = fmaxf(m, Ss[row * 224 + j + 8 * i]);
    #pragma unroll
    for (int off = 4; off; off >>= 1) m = fmaxf(m, __shfl_xor(m, off, 8));
    float s = 0.f;
    #pragma unroll
    for (int i = 0; i < 28; ++i) s += expf(Ss[row * 224 + j + 8 * i] - m);
    #pragma unroll
    for (int off = 4; off; off >>= 1) s += __shfl_xor(s, off, 8);
    float inv = 1.f / s;
    #pragma unroll
    for (int i = 0; i < 28; ++i) {
      int c = j + 8 * i;
      Ps[row * 224 + c] = (_Float16)(expf(Ss[row * 224 + c] - m) * inv);
    }
  }
  __syncthreads();

  // Phase 3: O = P V  (8 16x16 tiles, one per wave, 7 wmma over k=224)
  {
    int mt = wave >> 2, nt = wave & 3;
    v8f acc = {};
    #pragma unroll
    for (int kc = 0; kc < 7; ++kc) {
      v16h a  = frag_a(Ps + mt * 16 * 224 + kc * 32, 224);
      v16h bf = frag_b(Vs + nt * 16 * 224 + kc * 32, 224);
      acc = wmma16(a, bf, acc);
    }
    int mrow = (lane < 16) ? 0 : 8;
    int e = nt * 16 + (lane & 15);
    #pragma unroll
    for (int r = 0; r < 8; ++r) {
      int token = qt * 32 + mt * 16 + mrow + r;
      O[((size_t)b * NP + token) * DD + h * HD + e] = (_Float16)acc[r];
    }
  }
}

// ---------------------------------------------------------------- final head
__global__ void finalize_kernel(const float* __restrict__ LAT,
                                const float* __restrict__ ifp,
                                float* __restrict__ outSel,
                                float* __restrict__ outNum) {
  __shared__ float red[256];
  __shared__ float sc[KK];
  __shared__ int selid[TOPKN];
  int b = blockIdx.x, tid = threadIdx.x;
  {
    int k = tid >> 4, part = tid & 15;
    float s = 0.f;
    for (int i = part; i < TOKN; i += 16)
      s += LAT[((size_t)b * KK + k) * TOKN + i] * ifp[(size_t)b * TOKN + i];
    red[tid] = s;
  }
  __syncthreads();
  if (tid < KK) {
    float s = 0.f;
    #pragma unroll
    for (int j = 0; j < 16; ++j) s += red[tid * 16 + j];
    sc[tid] = s;
  }
  __syncthreads();
  if (tid == 0) {
    float aw[KK];
    float m = sc[0];
    for (int k = 1; k < KK; ++k) m = fmaxf(m, sc[k]);
    float sum = 0.f;
    for (int k = 0; k < KK; ++k) { aw[k] = expf(sc[k] - m); sum += aw[k]; }
    int cnt = 0;
    for (int k = 0; k < KK; ++k) { aw[k] /= sum; if (aw[k] > EPSV) cnt++; }
    int numr = cnt < 1 ? 1 : (cnt > TOPKN ? TOPKN : cnt);
    int used = 0, sidx[TOPKN], keys[TOPKN];
    for (int j = 0; j < TOPKN; ++j) {          // stable descending argsort, top-8
      int best = -1;
      for (int k = 0; k < KK; ++k)
        if (!(used & (1 << k)) && (best < 0 || aw[k] > aw[best])) best = k;
      used |= 1 << best; sidx[j] = best;
    }
    for (int j = 0; j < TOPKN; ++j) keys[j] = (j < numr) ? sidx[j] : (KK + j);
    for (int i = 0; i < TOPKN; ++i)            // argsort(keys) applied to sidx
      for (int j = 0; j < TOPKN - 1 - i; ++j)
        if (keys[j] > keys[j + 1]) {
          int t = keys[j]; keys[j] = keys[j + 1]; keys[j + 1] = t;
          t = sidx[j]; sidx[j] = sidx[j + 1]; sidx[j + 1] = t;
        }
    for (int j = 0; j < TOPKN; ++j) selid[j] = sidx[j];
    outNum[b] = (float)numr;
  }
  __syncthreads();
  for (int j = 0; j < TOPKN; ++j) {
    int id = selid[j];
    const float* src = LAT + ((size_t)b * KK + id) * TOKN;
    float v0 = src[tid], v1 = src[tid + 256];
    red[tid] = v0 * v0 + v1 * v1;
    __syncthreads();
    for (int off = 128; off; off >>= 1) {
      if (tid < off) red[tid] += red[tid + off];
      __syncthreads();
    }
    float inv = 1.f / fmaxf(sqrtf(red[0]), 1e-12f);
    __syncthreads();
    float* dst = outSel + ((size_t)b * TOPKN + j) * TOKN;
    dst[tid] = v0 * inv; dst[tid + 256] = v1 * inv;
  }
}

// ---------------------------------------------------------------- host launch
extern "C" void kernel_launch(void* const* d_in, const int* in_sizes, int n_in,
                              void* d_out, int out_size, void* d_ws, size_t ws_size,
                              hipStream_t stream) {
  const float* ifp   = (const float*)d_in[0];
  const float* patch = (const float*)d_in[1];
  const float* tpl   = (const float*)d_in[2];
  const float* ln1w  = (const float*)d_in[3];
  const float* ln1b  = (const float*)d_in[4];
  const float* qw    = (const float*)d_in[5];
  const float* qb    = (const float*)d_in[6];
  const float* kvw   = (const float*)d_in[7];
  const float* kvb   = (const float*)d_in[8];
  const float* pw    = (const float*)d_in[9];
  const float* pb    = (const float*)d_in[10];
  const float* ln2w  = (const float*)d_in[11];
  const float* ln2b  = (const float*)d_in[12];
  const float* f1w   = (const float*)d_in[13];
  const float* f1b   = (const float*)d_in[14];
  const float* f2w   = (const float*)d_in[15];
  const float* f2b   = (const float*)d_in[16];
  const float* afw   = (const float*)d_in[17];
  const float* afb   = (const float*)d_in[18];

  char* ws = (char*)d_ws;
  size_t off = 0;
  auto alloc = [&](size_t bytes) {
    size_t o = off; off = (off + bytes + 255) & ~(size_t)255; return o;
  };
  float*    X    = (float*)   (ws + alloc((size_t)MROWS * DD * 4));
  _Float16* B0   = (_Float16*)(ws + alloc((size_t)MROWS * DD * 2));
  _Float16* B1   = (_Float16*)(ws + alloc((size_t)MROWS * DD * 2));
  _Float16* B2   = (_Float16*)(ws + alloc((size_t)MROWS * 2 * DD * 2));
  _Float16* qw16 = (_Float16*)(ws + alloc((size_t)LL * DD * DD * 2));
  _Float16* kvw16= (_Float16*)(ws + alloc((size_t)LL * DD * 2 * DD * 2));
  _Float16* pw16 = (_Float16*)(ws + alloc((size_t)LL * DD * DD * 2));
  _Float16* f1w16= (_Float16*)(ws + alloc((size_t)LL * DD * 2 * DD * 2));
  _Float16* f2w16= (_Float16*)(ws + alloc((size_t)LL * 2 * DD * DD * 2));
  _Float16* afw16= (_Float16*)(ws + alloc((size_t)DD * TOKN * 2));
  _Float16* XT16 = (_Float16*)(ws + alloc((size_t)BB * KK * DD * 2));
  float*    LATp = (float*)   (ws + alloc((size_t)BB * KK * TOKN * 4));

  const int T = 256;
  auto cvt = [&](const float* s, _Float16* d, long n) {
    int g = (int)((n + T - 1) / T); if (g > 8192) g = 8192;
    cvt_f16_kernel<<<g, T, 0, stream>>>(s, d, n);
  };
  cvt(qw,  qw16,  (long)LL * DD * DD);
  cvt(kvw, kvw16, (long)LL * DD * 2 * DD);
  cvt(pw,  pw16,  (long)LL * DD * DD);
  cvt(f1w, f1w16, (long)LL * DD * 2 * DD);
  cvt(f2w, f2w16, (long)LL * 2 * DD * DD);
  cvt(afw, afw16, (long)DD * TOKN);

  initx_kernel<<<8192, T, 0, stream>>>(tpl, patch, X, (long)MROWS * DD);

  dim3 gQ (DD / 64,     MROWS / 128);
  dim3 gKV(2 * DD / 64, MROWS / 128);
  for (int l = 0; l < LL; ++l) {
    const _Float16* lqw  = qw16  + (size_t)l * DD * DD;
    const _Float16* lkvw = kvw16 + (size_t)l * DD * 2 * DD;
    const _Float16* lpw  = pw16  + (size_t)l * DD * DD;
    const _Float16* lf1  = f1w16 + (size_t)l * DD * 2 * DD;
    const _Float16* lf2  = f2w16 + (size_t)l * 2 * DD * DD;

    ln_kernel<<<MROWS, T, 0, stream>>>(X, ln1w + l * DD, ln1b + l * DD, B0);
    gemm_wmma_kernel<0><<<gQ,  T, 0, stream>>>(B0, DD, lqw,  DD,     qb  + l * DD,     B1, nullptr, DD,     DD);
    gemm_wmma_kernel<0><<<gKV, T, 0, stream>>>(B0, DD, lkvw, 2 * DD, kvb + l * 2 * DD, B2, nullptr, 2 * DD, DD);
    attn_kernel<<<dim3(NP / 32, HH, BB), T, 104448, stream>>>(B1, B2, B0);
    gemm_wmma_kernel<2><<<gQ,  T, 0, stream>>>(B0, DD, lpw,  DD,     pb  + l * DD,     nullptr, X, DD,     DD);
    ln_kernel<<<MROWS, T, 0, stream>>>(X, ln2w + l * DD, ln2b + l * DD, B1);
    gemm_wmma_kernel<1><<<gKV, T, 0, stream>>>(B1, DD,     lf1, 2 * DD, f1b + l * 2 * DD, B2, nullptr, 2 * DD, DD);
    gemm_wmma_kernel<2><<<gQ,  T, 0, stream>>>(B2, 2 * DD, lf2, DD,     f2b + l * DD,     nullptr, X, DD,     2 * DD);
  }

  {
    long n = (long)BB * KK * DD;
    extract_kernel<<<(int)((n + T - 1) / T), T, 0, stream>>>(X, XT16);
  }
  gemm_wmma_kernel<3><<<dim3(TOKN / 64, (BB * KK) / 128), T, 0, stream>>>(
      XT16, DD, afw16, TOKN, afb, nullptr, LATp, TOKN, DD);

  float* outSel = (float*)d_out;
  float* outNum = outSel + (size_t)BB * TOPKN * TOKN;
  finalize_kernel<<<BB, T, 0, stream>>>(LATp, ifp, outSel, outNum);
}
